// SymmetricChildEncoder_62148176773768
// MI455X (gfx1250) — matrix operand order repaired and checked by reading, not compiled, not run on hardware
//
#include <hip/hip_runtime.h>
#include <math.h>

// ---------------- problem constants ----------------
#define NNODES   20000
#define NEDGES   320000
#define HDIM     128
#define DIN_OP   162     // NUM_SEM + MAXPP + FS
#define DIN_EDGE 260     // 2*H + ETN
#define FGDIM    128
#define NTILES_M (NNODES/16)   // 1250

typedef __attribute__((ext_vector_type(16))) __bf16 v16bf;
typedef __attribute__((ext_vector_type(8)))  __bf16 v8bf;
typedef __attribute__((ext_vector_type(8)))  float  v8f;

__device__ __forceinline__ float leaky01(float x){ return x > 0.f ? x : 0.01f * x; }

// =====================================================================
// Pack W (dout=128 x din, row-major f32) into bf16 WMMA-B fragments.
// Fragment (kt,nt) lane L element j holds B[k][n] = W[n][k] with
//   n = nt*16 + (L&15),  k = kt*32 + (j>>3)*16 + (L>>4)*8 + (j&7)   (zero pad k>=din)
// Stored contiguously: out[((kt*8+nt)*32 + L)*16 + j]
// =====================================================================
__global__ void prep_pack_b(const float* __restrict__ W, int din, int ktiles,
                            __bf16* __restrict__ out)
{
    int idx = blockIdx.x * 256 + threadIdx.x;
    int total = ktiles * 8 * 512;
    if (idx >= total) return;
    int j    = idx & 15;
    int lane = (idx >> 4) & 31;
    int frag = idx >> 9;
    int nt = frag & 7, kt = frag >> 3;
    int n = nt * 16 + (lane & 15);
    int k = kt * 32 + (j >> 3) * 16 + (lane >> 4) * 8 + (j & 7);
    float v = (k < din) ? W[(long)n * din + k] : 0.f;
    out[idx] = (__bf16)v;
}

// =====================================================================
// Node GEMM: cf = (child_feats @ Wop^T + b) * exists   (20000x162 -> 20000x128)
// 8 waves/block, one 16-row tile per wave, K padded 162->192 (6 k-tiles).
// =====================================================================
__global__ void node_gemm(const float* __restrict__ feats, const float* __restrict__ exists,
                          const __bf16* __restrict__ pkB, const float* __restrict__ bias,
                          float* __restrict__ cfF, __bf16* __restrict__ cfB)
{
    int wave = threadIdx.x >> 5, lane = threadIdx.x & 31;
    int tile = blockIdx.x * 8 + wave;
    if (tile >= NTILES_M) return;
    int m = lane & 15, hi = lane >> 4;
    int rb = tile * 16;
    const float* ap = feats + (long)(rb + m) * DIN_OP;

    v8f acc[8];
    #pragma unroll
    for (int t = 0; t < 8; ++t) acc[t] = (v8f)(0.0f);

    #pragma unroll
    for (int kt = 0; kt < 6; ++kt) {
        v16bf a;
        #pragma unroll
        for (int j = 0; j < 16; ++j) {
            int k = kt * 32 + ((j >> 3) * 16) + hi * 8 + (j & 7);
            a[j] = (k < DIN_OP) ? (__bf16)ap[k] : (__bf16)0.f;
        }
        #pragma unroll
        for (int t = 0; t < 8; ++t) {
            v16bf b = *(const v16bf*)(pkB + (((kt * 8 + t) * 32 + lane) << 4));
            acc[t] = __builtin_amdgcn_wmma_f32_16x16x32_bf16(false, a, false, b,
                                                             (short)0, acc[t], false, false);
        }
    }
    #pragma unroll
    for (int r = 0; r < 8; ++r) {
        int orow = rb + hi * 8 + r;
        float ex = exists[orow];
        #pragma unroll
        for (int t = 0; t < 8; ++t) {
            int n = t * 16 + m;
            float v = (acc[t][r] + bias[n]) * ex;
            cfF[(long)orow * HDIM + n] = v;
            cfB[(long)orow * HDIM + n] = (__bf16)v;
        }
    }
}

// =====================================================================
// Edge message GEMM fused with scatter-max.
// msg = leaky([cf[from] | cf[to] | etype] @ We^T + be)  (320000x260 -> x128)
// then cfnew[from] = elementwise-max.  K padded 260->288 (9 k-tiles).
// cfnew zero-initialized; values scattered via int atomicMax (valid since
// the result is max(0, x): positive-float order == signed-int-bit order).
// =====================================================================
__global__ void edge_gemm_scatter(const __bf16* __restrict__ cf, const float* __restrict__ etype,
                                  const int* __restrict__ eidx, const __bf16* __restrict__ pkB,
                                  const float* __restrict__ bias, float* __restrict__ cfnew)
{
    int wave = threadIdx.x >> 5, lane = threadIdx.x & 31;
    int m = lane & 15, hi = lane >> 4;
    int eb = (blockIdx.x * 8 + wave) * 16;
    int e = eb + m;
    int nfrom = eidx[2 * e];
    int nto   = eidx[2 * e + 1];
    const __bf16* pf = cf + (long)nfrom * HDIM;
    const __bf16* pt = cf + (long)nto   * HDIM;

    v8f acc[8];
    #pragma unroll
    for (int t = 0; t < 8; ++t) acc[t] = (v8f)(0.0f);

    #pragma unroll
    for (int kt = 0; kt < 9; ++kt) {
        v16bf a;
        if (kt < 8) {
            const __bf16* p = (kt < 4) ? pf : pt;
            int kb = (kt & 3) * 32;
            v8bf lo  = *(const v8bf*)(p + kb + hi * 8);
            v8bf hi8 = *(const v8bf*)(p + kb + 16 + hi * 8);
            #pragma unroll
            for (int j = 0; j < 8; ++j) { a[j] = lo[j]; a[8 + j] = hi8[j]; }
        } else {
            #pragma unroll
            for (int j = 0; j < 16; ++j) a[j] = (__bf16)0.f;
            if (hi == 0) {     // K-local 0..3 live only in low lane half, elems 0..3
                #pragma unroll
                for (int j = 0; j < 4; ++j) a[j] = (__bf16)etype[4 * e + j];
            }
        }
        #pragma unroll
        for (int t = 0; t < 8; ++t) {
            v16bf b = *(const v16bf*)(pkB + (((kt * 8 + t) * 32 + lane) << 4));
            acc[t] = __builtin_amdgcn_wmma_f32_16x16x32_bf16(false, a, false, b,
                                                             (short)0, acc[t], false, false);
        }
    }
    #pragma unroll
    for (int r = 0; r < 8; ++r) {
        int orow = eb + hi * 8 + r;
        int dst = eidx[2 * orow];
        #pragma unroll
        for (int t = 0; t < 8; ++t) {
            int n = t * 16 + m;
            float v = leaky01(acc[t][r] + bias[n]);
            atomicMax((int*)(cfnew + (long)dst * HDIM + n), __float_as_int(v));
        }
    }
}

// =====================================================================
// Fused skip / geo GEMMs; keep only per-column max over all nodes.
// =====================================================================
__global__ void geo_gemm_max(const float* __restrict__ geo, const float* __restrict__ exists,
                             const __bf16* __restrict__ pkS, const float* __restrict__ skb,
                             const __bf16* __restrict__ pkG, const float* __restrict__ cgb,
                             float* __restrict__ partS, float* __restrict__ partG)
{
    __shared__ float smS[8 * 128];
    __shared__ float smG[8 * 128];
    int wave = threadIdx.x >> 5, lane = threadIdx.x & 31;
    int m = lane & 15, hi = lane >> 4;
    int tile = blockIdx.x * 8 + wave;

    float ms[8], mg[8];
    #pragma unroll
    for (int t = 0; t < 8; ++t) { ms[t] = -INFINITY; mg[t] = -INFINITY; }

    if (tile < NTILES_M) {
        int rb = tile * 16;
        const float* ap = geo + (long)(rb + m) * FGDIM;
        v8f accS[8], accG[8];
        #pragma unroll
        for (int t = 0; t < 8; ++t) { accS[t] = (v8f)(0.0f); accG[t] = (v8f)(0.0f); }
        #pragma unroll
        for (int kt = 0; kt < 4; ++kt) {
            v16bf a;
            #pragma unroll
            for (int j = 0; j < 16; ++j) {
                int k = kt * 32 + ((j >> 3) * 16) + hi * 8 + (j & 7);
                a[j] = (__bf16)ap[k];
            }
            #pragma unroll
            for (int t = 0; t < 8; ++t) {
                v16bf bS = *(const v16bf*)(pkS + (((kt * 8 + t) * 32 + lane) << 4));
                v16bf bG = *(const v16bf*)(pkG + (((kt * 8 + t) * 32 + lane) << 4));
                accS[t] = __builtin_amdgcn_wmma_f32_16x16x32_bf16(false, a, false, bS,
                                                                  (short)0, accS[t], false, false);
                accG[t] = __builtin_amdgcn_wmma_f32_16x16x32_bf16(false, a, false, bG,
                                                                  (short)0, accG[t], false, false);
            }
        }
        #pragma unroll
        for (int r = 0; r < 8; ++r) {
            int orow = rb + hi * 8 + r;
            float ex = exists[orow];
            #pragma unroll
            for (int t = 0; t < 8; ++t) {
                int n = t * 16 + m;
                float vs = (accS[t][r] + skb[n]) * ex;
                float vg = (accG[t][r] + cgb[n]) * ex;
                ms[t] = fmaxf(ms[t], vs);
                mg[t] = fmaxf(mg[t], vg);
            }
        }
    }
    // combine lane halves (rows r and r+8 live in opposite halves, same n)
    #pragma unroll
    for (int t = 0; t < 8; ++t) {
        ms[t] = fmaxf(ms[t], __shfl_xor(ms[t], 16, 32));
        mg[t] = fmaxf(mg[t], __shfl_xor(mg[t], 16, 32));
    }
    if (hi == 0) {
        #pragma unroll
        for (int t = 0; t < 8; ++t) {
            smS[wave * 128 + t * 16 + m] = ms[t];
            smG[wave * 128 + t * 16 + m] = mg[t];
        }
    }
    __syncthreads();
    if (threadIdx.x < 128) {
        int n = threadIdx.x;
        float a = -INFINITY, b = -INFINITY;
        #pragma unroll
        for (int w = 0; w < 8; ++w) {
            a = fmaxf(a, smS[w * 128 + n]);
            b = fmaxf(b, smG[w * 128 + n]);
        }
        partS[(long)blockIdx.x * 128 + n] = a;
        partG[(long)blockIdx.x * 128 + n] = b;
    }
}

// ---------------- utility kernels ----------------
__global__ void zero_f32_k(float* __restrict__ p, long n)
{
    long i = (long)blockIdx.x * 256 + threadIdx.x;
    if (i < n) p[i] = 0.f;
}
__global__ void f32_to_bf16_k(const float* __restrict__ s, __bf16* __restrict__ d, long n)
{
    long i = (long)blockIdx.x * 256 + threadIdx.x;
    if (i < n) d[i] = (__bf16)s[i];
}
__global__ void colmax_partial(const float* __restrict__ tab, float* __restrict__ part, int rowsPerBlk)
{
    int n = threadIdx.x;                    // 128 threads, fully coalesced per row
    long r0 = (long)blockIdx.x * rowsPerBlk;
    float mx = -INFINITY;
    for (int r = 0; r < rowsPerBlk; ++r) mx = fmaxf(mx, tab[(r0 + r) * HDIM + n]);
    part[(long)blockIdx.x * HDIM + n] = mx;
}
__global__ void colmax_final(const float* __restrict__ part, int nblocks, float* __restrict__ out)
{
    int n = threadIdx.x;
    float mx = -INFINITY;
    for (int b = 0; b < nblocks; ++b) mx = fmaxf(mx, part[(long)b * HDIM + n]);
    out[n] = mx;
}

// parent_feat = leaky(concat(ipf0,ipf1,ipf2) @ second_w^T + second_b)
__global__ void parent_final(const float* __restrict__ ipf, const float* __restrict__ W,
                             const float* __restrict__ b, float* __restrict__ out)
{
    __shared__ float cat[384];
    int n = threadIdx.x;
    cat[n] = ipf[n]; cat[128 + n] = ipf[128 + n]; cat[256 + n] = ipf[256 + n];
    __syncthreads();
    float d = b[n];
    for (int k = 0; k < 384; ++k) d += cat[k] * W[(long)n * 384 + k];
    out[n] = leaky01(d);
}

// parent_geo = leaky(sg + groupnorm(pg @ second_geo_w^T + b) * gnw + gnb)
__global__ void geo_final(const float* __restrict__ partS, const float* __restrict__ partG, int nb,
                          const float* __restrict__ W, const float* __restrict__ b2,
                          const float* __restrict__ gnw, const float* __restrict__ gnb,
                          float* __restrict__ out)
{
    __shared__ float pg[128];
    __shared__ float yv[128];
    int n = threadIdx.x;
    float msk = -INFINITY, mgm = -INFINITY;
    for (int w = 0; w < nb; ++w) {
        msk = fmaxf(msk, partS[(long)w * 128 + n]);
        mgm = fmaxf(mgm, partG[(long)w * 128 + n]);
    }
    float sg = leaky01(msk);
    pg[n] = leaky01(mgm);
    __syncthreads();
    float y = b2[n];
    for (int k = 0; k < 128; ++k) y += pg[k] * W[(long)n * 128 + k];
    yv[n] = y;
    __syncthreads();
    int g = n >> 3;                 // 16 groups of 8 channels
    float mu = 0.f;
    #pragma unroll
    for (int i = 0; i < 8; ++i) mu += yv[g * 8 + i];
    mu *= 0.125f;
    float var = 0.f;
    #pragma unroll
    for (int i = 0; i < 8; ++i) { float d = yv[g * 8 + i] - mu; var += d * d; }
    var *= 0.125f;
    float xn = (y - mu) * rsqrtf(var + 1e-5f);
    out[128 + n] = leaky01(sg + xn * gnw[n] + gnb[n]);
}

// =====================================================================
extern "C" void kernel_launch(void* const* d_in, const int* in_sizes, int n_in,
                              void* d_out, int out_size, void* d_ws, size_t ws_size,
                              hipStream_t stream)
{
    (void)in_sizes; (void)n_in; (void)out_size; (void)ws_size;
    const float* child_feats  = (const float*)d_in[0];
    const float* child_geo    = (const float*)d_in[1];
    const float* child_exists = (const float*)d_in[2];
    const float* etype        = (const float*)d_in[3];
    const int*   eidx         = (const int*)d_in[4];
    const float* cop_w  = (const float*)d_in[5];
    const float* cop_b  = (const float*)d_in[6];
    const float* sec_w  = (const float*)d_in[7];
    const float* sec_b  = (const float*)d_in[8];
    const float* edge_w = (const float*)d_in[9];
    const float* edge_b = (const float*)d_in[10];
    const float* cgeo_w = (const float*)d_in[11];
    const float* cgeo_b = (const float*)d_in[12];
    const float* sgeo_w = (const float*)d_in[13];
    const float* sgeo_b = (const float*)d_in[14];
    const float* gn_w   = (const float*)d_in[15];
    const float* gn_b   = (const float*)d_in[16];
    const float* skip_w = (const float*)d_in[17];
    const float* skip_b = (const float*)d_in[18];

    char* ws = (char*)d_ws;
    size_t o = 0;
    __bf16* cfB  = (__bf16*)(ws + o); o += (size_t)NNODES * HDIM * 2;   // 5.12 MB
    float*  cfF  = (float*)(ws + o);  o += (size_t)NNODES * HDIM * 4;   // 10.24 MB
    __bf16* pkCO = (__bf16*)(ws + o); o += (size_t)48  * 512 * 2;       // 6 kt x 8 nt
    __bf16* pkE  = (__bf16*)(ws + o); o += (size_t)144 * 512 * 2;       // 2 x 9 x 8
    __bf16* pkS  = (__bf16*)(ws + o); o += (size_t)32  * 512 * 2;
    __bf16* pkG  = (__bf16*)(ws + o); o += (size_t)32  * 512 * 2;
    float*  ipf  = (float*)(ws + o);  o += 3 * 128 * 4;
    float*  cmp  = (float*)(ws + o);  o += 100 * 128 * 4;
    float*  pS   = (float*)(ws + o);  o += 157 * 128 * 4;
    float*  pG   = (float*)(ws + o);  o += 157 * 128 * 4;

    // --- pack weights into WMMA B fragments (bf16) ---
    prep_pack_b<<<(6 * 8 * 512 + 255) / 256, 256, 0, stream>>>(cop_w,  DIN_OP,   6, pkCO);
    prep_pack_b<<<(9 * 8 * 512 + 255) / 256, 256, 0, stream>>>(edge_w, DIN_EDGE, 9, pkE);
    prep_pack_b<<<(9 * 8 * 512 + 255) / 256, 256, 0, stream>>>(edge_w + 128 * 260, DIN_EDGE, 9, pkE + 72 * 512);
    prep_pack_b<<<(4 * 8 * 512 + 255) / 256, 256, 0, stream>>>(skip_w, FGDIM, 4, pkS);
    prep_pack_b<<<(4 * 8 * 512 + 255) / 256, 256, 0, stream>>>(cgeo_w, FGDIM, 4, pkG);

    // --- node GEMM: cf0 ---
    node_gemm<<<(NTILES_M + 7) / 8, 256, 0, stream>>>(child_feats, child_exists, pkCO, cop_b, cfF, cfB);
    colmax_partial<<<100, 128, 0, stream>>>(cfF, cmp, NNODES / 100);
    colmax_final<<<1, 128, 0, stream>>>(cmp, 100, ipf);

    // --- message-passing iterations ---
    const long tabN = (long)NNODES * HDIM;
    for (int it = 0; it < 2; ++it) {
        zero_f32_k<<<(int)((tabN + 255) / 256), 256, 0, stream>>>(cfF, tabN);
        edge_gemm_scatter<<<NEDGES / 128, 256, 0, stream>>>(cfB, etype, eidx,
                                                            pkE + it * 72 * 512,
                                                            edge_b + it * 128, cfF);
        colmax_partial<<<100, 128, 0, stream>>>(cfF, cmp, NNODES / 100);
        colmax_final<<<1, 128, 0, stream>>>(cmp, 100, ipf + (it + 1) * 128);
        f32_to_bf16_k<<<(int)((tabN + 255) / 256), 256, 0, stream>>>(cfF, cfB, tabN);
    }

    // --- parent semantic head ---
    parent_final<<<1, 128, 0, stream>>>(ipf, sec_w, sec_b, (float*)d_out);

    // --- geometry path ---
    geo_gemm_max<<<(NTILES_M + 7) / 8, 256, 0, stream>>>(child_geo, child_exists,
                                                         pkS, skip_b, pkG, cgeo_b, pS, pG);
    geo_final<<<1, 128, 0, stream>>>(pS, pG, (NTILES_M + 7) / 8, sgeo_w, sgeo_b,
                                     gn_w, gn_b, (float*)d_out);
}
